// SelfAttention_5085241279142
// MI455X (gfx1250) — compile-verified, matrix operand order checked by reading, not run on hardware
//
#include <hip/hip_runtime.h>

// ---- CDNA5 WMMA types --------------------------------------------------
typedef __attribute__((ext_vector_type(16))) __bf16 v16bf;
typedef __attribute__((ext_vector_type(8)))  float  v8f;

#define BATCH 4
#define SEQ   2048
#define CH    512
#define CQK   64
#define KTOT  1536            // 3*CH (conv taps unrolled into K)
#define NOUT  640             // CQK + CQK + CH fused projection width

// workspace layout (bytes)
#define OFF_WT  0u
#define SZ_WT   ((size_t)NOUT * KTOT * 2)                 // 1,966,080
#define OFF_Q   (OFF_WT + SZ_WT)
#define SZ_QK   ((size_t)BATCH * SEQ * CQK * 2)           // 1,048,576
#define OFF_K   (OFF_Q + SZ_QK)
#define OFF_VT  (OFF_K + SZ_QK)                           // V stored transposed [C][N]

__device__ __forceinline__ v8f wmma_bf16(v16bf a, v16bf b, v8f c) {
  return __builtin_amdgcn_wmma_f32_16x16x32_bf16(false, a, false, b, (short)0, c,
                                                 false, false);
}

__device__ __forceinline__ v16bf pack2x16B(const __bf16* p0, const __bf16* p1) {
  union { uint4 u[2]; v16bf v; } r;
  r.u[0] = *(const uint4*)p0;
  r.u[1] = *(const uint4*)p1;
  return r.v;
}

// ---- Kernel 1: transpose + bf16-convert fused weight [640][1536] -------
__global__ void prep_weights(const float* __restrict__ Wq,
                             const float* __restrict__ Wk,
                             const float* __restrict__ Wv,
                             __bf16* __restrict__ Wt) {
  int idx = blockIdx.x * blockDim.x + threadIdx.x;
  if (idx >= NOUT * KTOT) return;
  int co = idx / KTOT;
  int kk = idx - co * KTOT;              // kk = t*CH + ci ; W flat = kk*Cout + co
  float v;
  if (co < CQK)          v = Wq[(size_t)kk * CQK + co];
  else if (co < 2 * CQK) v = Wk[(size_t)kk * CQK + (co - CQK)];
  else                   v = Wv[(size_t)kk * CH  + (co - 2 * CQK)];
  Wt[(size_t)co * KTOT + kk] = (__bf16)v;
}

// ---- Kernel 2: fused QKV conv-as-GEMM  Y[8192,640] = Xshift @ Wt^T -----
__global__ __launch_bounds__(256)
void qkv_gemm(const float* __restrict__ x, const __bf16* __restrict__ Wt,
              const float* __restrict__ bq, const float* __restrict__ bk,
              const float* __restrict__ bv,
              __bf16* __restrict__ qh, __bf16* __restrict__ kh,
              __bf16* __restrict__ vT) {
  const int lane = threadIdx.x & 31;
  const int wave = threadIdx.x >> 5;
  const int tile  = blockIdx.x * 8 + wave;
  const int mtile = tile / (NOUT / 16);
  const int ctile = tile - mtile * (NOUT / 16);
  const int sel  = (lane >= 16) ? 8 : 0;   // K half-select per WMMA 16-bit layout
  const int rit  = lane & 15;
  const int grow = mtile * 16 + rit;       // A-matrix row for this lane
  const int b    = grow >> 11;
  const int nn   = grow & 2047;
  const int col  = ctile * 16 + rit;       // B/C/D column for this lane
  const __bf16* wrow = Wt + (size_t)col * KTOT;

  v8f acc = {0.f, 0.f, 0.f, 0.f, 0.f, 0.f, 0.f, 0.f};
  for (int kk = 0; kk < KTOT; kk += 32) {
    const int t   = kk >> 9;               // conv tap
    const int pos = nn - 1 + t;            // SAME padding
    const bool valid = (pos >= 0) && (pos < SEQ);
    const int ci  = (kk & 511) + sel;
    union { __bf16 h[16]; v16bf v; } ra;
    if (valid) {
      const float* pa = x + ((size_t)(b * SEQ + pos)) * CH + ci;
      alignas(16) float tf[16];
      ((float4*)tf)[0] = *(const float4*)(pa);
      ((float4*)tf)[1] = *(const float4*)(pa + 4);
      ((float4*)tf)[2] = *(const float4*)(pa + 16);
      ((float4*)tf)[3] = *(const float4*)(pa + 20);
#pragma unroll
      for (int i = 0; i < 16; ++i) ra.h[i] = (__bf16)tf[i];
    } else {
#pragma unroll
      for (int i = 0; i < 16; ++i) ra.h[i] = (__bf16)0.0f;
    }
    v16bf bmat = pack2x16B(wrow + kk + sel, wrow + kk + sel + 16);
    acc = wmma_bf16(ra.v, bmat, acc);
  }

  const float bias = (col < CQK) ? bq[col]
                   : (col < 2 * CQK) ? bk[col - CQK] : bv[col - 2 * CQK];
#pragma unroll
  for (int r = 0; r < 8; ++r) {
    const int row = mtile * 16 + r + ((lane >= 16) ? 8 : 0);
    const int bb = row >> 11, n2 = row & 2047;
    const float val = acc[r] + bias;
    if (col < CQK)
      qh[((size_t)(bb * SEQ + n2)) * CQK + col] = (__bf16)val;
    else if (col < 2 * CQK)
      kh[((size_t)(bb * SEQ + n2)) * CQK + (col - CQK)] = (__bf16)val;
    else
      vT[((size_t)(bb * CH + (col - 2 * CQK))) * SEQ + n2] = (__bf16)val;  // transposed V
  }
}

// ---- Kernel 3: fused scores + softmax + beta write + beta@V + residual --
__global__ __launch_bounds__(256)
void attention(const float* __restrict__ x, const __bf16* __restrict__ qh,
               const __bf16* __restrict__ kh, const __bf16* __restrict__ vT,
               const float* __restrict__ gamma_p, float* __restrict__ out) {
  __shared__ __bf16 Plds[16 * SEQ];           // 64 KB: P row-block (bf16)
  const int lane = threadIdx.x & 31;
  const int wave = threadIdx.x >> 5;
  const int b     = blockIdx.x >> 7;          // /128
  const int nbase = (blockIdx.x & 127) * 16;
  const int sel = (lane >= 16) ? 8 : 0;
  const int rit = lane & 15;
  const float g = gamma_p[0];
  float* beta = out + (size_t)BATCH * SEQ * CH;

  // Load Q A-tiles for this row block once (K = 64 -> two k32 steps).
  const __bf16* qrow = qh + ((size_t)(b * SEQ + nbase + rit)) * CQK;
  const v16bf aq0 = pack2x16B(qrow + sel,      qrow + 16 + sel);
  const v16bf aq1 = pack2x16B(qrow + 32 + sel, qrow + 48 + sel);

  float mrun[8], srun[8];
#pragma unroll
  for (int r = 0; r < 8; ++r) { mrun[r] = -1e30f; srun[r] = 0.f; }

  // ---- Sweep 1: per-wave online softmax stats over its 256-col strip ---
  for (int j = 0; j < 16; ++j) {
    const int mbase = wave * 256 + j * 16;
    const __bf16* krow = kh + ((size_t)(b * SEQ + mbase + rit)) * CQK;
    v16bf bk0 = pack2x16B(krow + sel,      krow + 16 + sel);
    v16bf bk1 = pack2x16B(krow + 32 + sel, krow + 48 + sel);
    v8f acc = {0.f, 0.f, 0.f, 0.f, 0.f, 0.f, 0.f, 0.f};
    acc = wmma_bf16(aq0, bk0, acc);
    acc = wmma_bf16(aq1, bk1, acc);
#pragma unroll
    for (int r = 0; r < 8; ++r) {
      float v = acc[r];
      float tm = v;                           // row-max across the 16 cols
      tm = fmaxf(tm, __shfl_xor(tm, 1, 32));
      tm = fmaxf(tm, __shfl_xor(tm, 2, 32));
      tm = fmaxf(tm, __shfl_xor(tm, 4, 32));
      tm = fmaxf(tm, __shfl_xor(tm, 8, 32));
      const float mnew = fmaxf(mrun[r], tm);
      float ts = __expf(v - mnew);            // row-sum of exp across 16 cols
      ts += __shfl_xor(ts, 1, 32);
      ts += __shfl_xor(ts, 2, 32);
      ts += __shfl_xor(ts, 4, 32);
      ts += __shfl_xor(ts, 8, 32);
      srun[r] = srun[r] * __expf(mrun[r] - mnew) + ts;
      mrun[r] = mnew;
    }
  }

  // ---- Cross-wave reduce of (m,s) through LDS (reuse Plds space) -------
  float* red = (float*)Plds;                  // [0..127]=m, [128..255]=s
  if (lane == 0 || lane == 16) {
#pragma unroll
    for (int r = 0; r < 8; ++r) {
      const int row = r + ((lane == 16) ? 8 : 0);
      red[wave * 16 + row] = mrun[r];
      red[128 + wave * 16 + row] = srun[r];
    }
  }
  __syncthreads();
  float Mf[8], Sinv[8];
#pragma unroll
  for (int r = 0; r < 8; ++r) {
    const int row = r + sel;
    float M = -1e30f;
    for (int w = 0; w < 8; ++w) M = fmaxf(M, red[w * 16 + row]);
    float S = 0.f;
    for (int w = 0; w < 8; ++w) S += red[128 + w * 16 + row] * __expf(red[w * 16 + row] - M);
    Mf[r] = M;
    Sinv[r] = 1.f / S;
  }
  __syncthreads();

  // ---- Sweep 2: recompute scores, write normalized beta, stash P bf16 --
  for (int j = 0; j < 16; ++j) {
    const int mbase = wave * 256 + j * 16;
    const __bf16* krow = kh + ((size_t)(b * SEQ + mbase + rit)) * CQK;
    v16bf bk0 = pack2x16B(krow + sel,      krow + 16 + sel);
    v16bf bk1 = pack2x16B(krow + 32 + sel, krow + 48 + sel);
    v8f acc = {0.f, 0.f, 0.f, 0.f, 0.f, 0.f, 0.f, 0.f};
    acc = wmma_bf16(aq0, bk0, acc);
    acc = wmma_bf16(aq1, bk1, acc);
    const int colm = mbase + rit;
#pragma unroll
    for (int r = 0; r < 8; ++r) {
      const int row = r + sel;
      const float p = __expf(acc[r] - Mf[r]) * Sinv[r];
      beta[((size_t)(b * SEQ + nbase + row)) * SEQ + colm] = p;
      Plds[row * SEQ + colm] = (__bf16)p;
    }
  }
  __syncthreads();

  // ---- Sweep 3: O[16,512] = P[16,2048] @ V ; V read transposed ---------
  v8f oacc[4];
#pragma unroll
  for (int j = 0; j < 4; ++j)
    oacc[j] = (v8f){0.f, 0.f, 0.f, 0.f, 0.f, 0.f, 0.f, 0.f};
  for (int kk = 0; kk < SEQ; kk += 32) {
    const __bf16* pp = &Plds[rit * SEQ + kk + sel];
    v16bf ap = pack2x16B(pp, pp + 16);        // ds_load_b128 x2
#pragma unroll
    for (int j = 0; j < 4; ++j) {
      const int ch = (wave * 4 + j) * 16 + rit;
      const __bf16* pv = vT + ((size_t)(b * CH + ch)) * SEQ + kk + sel;
      oacc[j] = wmma_bf16(ap, pack2x16B(pv, pv + 16), oacc[j]);
    }
  }
#pragma unroll
  for (int j = 0; j < 4; ++j) {
    const int col = (wave * 4 + j) * 16 + rit;
#pragma unroll
    for (int r = 0; r < 8; ++r) {
      const int row = nbase + r + sel;
      const size_t idx = ((size_t)(b * SEQ + row)) * CH + col;
      out[idx] = g * oacc[j][r] + x[idx];
    }
  }
  if (blockIdx.x == 0 && threadIdx.x == 0)
    out[(size_t)BATCH * SEQ * CH + (size_t)BATCH * SEQ * SEQ] = g;  // gamma
}

// ------------------------------------------------------------------------
extern "C" void kernel_launch(void* const* d_in, const int* in_sizes, int n_in,
                              void* d_out, int out_size, void* d_ws, size_t ws_size,
                              hipStream_t stream) {
  (void)in_sizes; (void)n_in; (void)out_size; (void)ws_size;
  const float* x  = (const float*)d_in[0];
  const float* Wq = (const float*)d_in[1];
  const float* bq = (const float*)d_in[2];
  const float* Wk = (const float*)d_in[3];
  const float* bk = (const float*)d_in[4];
  const float* Wv = (const float*)d_in[5];
  const float* bv = (const float*)d_in[6];
  const float* gm = (const float*)d_in[7];
  float* out = (float*)d_out;
  char* ws = (char*)d_ws;
  __bf16* Wt = (__bf16*)(ws + OFF_WT);
  __bf16* qh = (__bf16*)(ws + OFF_Q);
  __bf16* kh = (__bf16*)(ws + OFF_K);
  __bf16* vT = (__bf16*)(ws + OFF_VT);

  prep_weights<<<(NOUT * KTOT + 255) / 256, 256, 0, stream>>>(Wq, Wk, Wv, Wt);
  // 8192/16 row tiles * 640/16 col tiles = 20480 wave-tiles / 8 waves per block
  qkv_gemm<<<2560, 256, 0, stream>>>(x, Wt, bq, bk, bv, qh, kh, vT);
  attention<<<BATCH * (SEQ / 16), 256, 0, stream>>>(x, qh, kh, vT, gm, out);
}
